// HybridEEGGraphModel_11742440587823
// MI455X (gfx1250) — compile-verified, matrix operand order checked by reading, not run on hardware
//
#include <hip/hip_runtime.h>
#include <hip/hip_bf16.h>
#include <stdint.h>

// ---------------------------------------------------------------------------
// HybridEEGGraphModel for MI455X (gfx1250, wave32, WMMA).
// Dominant cost: multi-scale conv1d (~46 GFLOP) -> v_wmma_f32_16x16x32_f16.
// Staging uses GLOBAL_LOAD_ASYNC_TO_LDS_* (ASYNCcnt) where available.
// ---------------------------------------------------------------------------

typedef __attribute__((ext_vector_type(16))) _Float16 v16h;
typedef __attribute__((ext_vector_type(8)))  float    v8f;
typedef __attribute__((ext_vector_type(4)))  int      int4v;

#define XPAD_W   2240      // 100 left pad + 2000 + right slack (zeroed)
#define TPRIME   2001
#define INV_BN   0.999995f // 1/sqrt(1 + 1e-5)

#if defined(__has_builtin)
#  if __has_builtin(__builtin_amdgcn_global_load_async_to_lds_b128) && \
      __has_builtin(__builtin_amdgcn_global_load_async_to_lds_b32)  && \
      __has_builtin(__builtin_amdgcn_s_wait_asynccnt)
#    define HAVE_ASYNC_LDS 1
#  endif
#endif

#ifdef HAVE_ASYNC_LDS
// Builtin signatures (from toolchain diagnostics):
//   b128: (AS1 int4v*, AS3 int4v*, imm offset, imm cpol)
//   b32 : (AS1 int*,   AS3 int*,   imm offset, imm cpol)
typedef __attribute__((address_space(1))) int4v g_int4v;
typedef __attribute__((address_space(3))) int4v l_int4v;
typedef __attribute__((address_space(1))) int   g_int;
typedef __attribute__((address_space(3))) int   l_int;
__device__ __forceinline__ void async_cp_b128(const void* g, void* l) {
  __builtin_amdgcn_global_load_async_to_lds_b128((g_int4v*)g, (l_int4v*)l, 0, 0);
}
__device__ __forceinline__ void async_cp_b32(const void* g, void* l) {
  __builtin_amdgcn_global_load_async_to_lds_b32((g_int*)g, (l_int*)l, 0, 0);
}
__device__ __forceinline__ void async_wait_all() { __builtin_amdgcn_s_wait_asynccnt(0); }
#endif

__device__ __forceinline__ float geluf(float x)   { return 0.5f * x * (1.0f + erff(x * 0.70710678118f)); }
__device__ __forceinline__ float sigmf(float x)   { return 1.0f / (1.0f + expf(-x)); }
__device__ __forceinline__ float eluf(float x)    { return x > 0.0f ? x : (expf(x) - 1.0f); }
// 16-bit WMMA fragment K mapping (per ISA 7.12.2, 16-bit A 16x32 layout)
__device__ __forceinline__ int kmap(int e, int lane) {
  int b = (lane & 16) ? 8 : 0;
  return (e < 8) ? (e + b) : (e + 8 + b);
}

// ---------------------------------------------------------------------------
// Pack x (fp32 [N,2000]) into zero-padded f16 rows [N, 2240], data at col 100.
// ---------------------------------------------------------------------------
__global__ void pack_x_kernel(const float* __restrict__ x, _Float16* __restrict__ xpad) {
  const int n = blockIdx.x;
  for (int i = threadIdx.x; i < XPAD_W; i += blockDim.x) {
    float v = (i >= 100 && i < 2100) ? x[(size_t)n * 2000 + (i - 100)] : 0.0f;
    xpad[(size_t)n * XPAD_W + i] = (_Float16)v;
  }
}

// ---------------------------------------------------------------------------
// Pack conv weights into WMMA A-fragment order (f16, zero-padded K).
//  wms: 3 scales -> (2+4+7) K-chunks x 2 M-halves = 26 frags of 32 lanes x 16 halves
//  wsp: spatial conv: 64 c-chunks x 2 M-halves = 128 frags
// ---------------------------------------------------------------------------
__global__ void pack_w_kernel(const float* __restrict__ w50, const float* __restrict__ w100,
                              const float* __restrict__ w200, const float* __restrict__ w2d,
                              _Float16* __restrict__ wms, _Float16* __restrict__ wsp) {
  const int KS[3] = {50, 100, 200};
  const int FB[3] = {0, 4, 12};
  const float* WP[3] = {w50, w100, w200};
  for (int idx = threadIdx.x; idx < 26 * 512; idx += blockDim.x) {
    int f = idx >> 9, lane = (idx >> 4) & 31, e = idx & 15;
    int s = (f < 4) ? 0 : ((f < 12) ? 1 : 2);
    int r = f - FB[s], c = r >> 1, m = r & 1;
    int row = m * 16 + (lane & 15);
    int K = c * 32 + kmap(e, lane);
    float v = (K < KS[s]) ? WP[s][row * KS[s] + K] : 0.0f;
    wms[idx] = (_Float16)v;
  }
  for (int idx = threadIdx.x; idx < 128 * 512; idx += blockDim.x) {
    int f = idx >> 9, lane = (idx >> 4) & 31, e = idx & 15;
    int c = f >> 1, m = f & 1;
    int o = m * 16 + (lane & 15);
    int K = kmap(e, lane);
    float v = (K < 20) ? w2d[((size_t)o * 64 + c) * 20 + K] : 0.0f;
    wsp[idx] = (_Float16)v;
  }
}

// ---------------------------------------------------------------------------
// Multi-scale conv1d + BN + GELU + time-mean, fused. Output s[N,32].
// Block = one series n, 8 waves; x row AND packed weights staged in LDS, so
// the steady-state inner loop is ds_load + v_wmma only.
// ---------------------------------------------------------------------------
__global__ void __launch_bounds__(256) conv_ms_kernel(
    const _Float16* __restrict__ xpad, const _Float16* __restrict__ wfrag,
    const float* __restrict__ bn_g, const float* __restrict__ bn_b,
    float* __restrict__ s_out) {
  const int n    = blockIdx.x;
  const int tid  = threadIdx.x;
  const int wave = tid >> 5;
  const int lane = tid & 31;
  const int tcol = lane & 15;
  const int hi   = (lane & 16) ? 8 : 0;

  __shared__ __align__(32) _Float16 xrow[XPAD_W];          // 4480 B
  __shared__ __align__(32) _Float16 wlds[26 * 512];        // 26624 B
  __shared__ float sg[96], sb[96], s_lds[32];

#ifdef HAVE_ASYNC_LDS
  {
    const char* gx = (const char*)(xpad + (size_t)n * XPAD_W);
    char* lx = (char*)xrow;
    for (int i = tid; i < XPAD_W * 2 / 16; i += 256)       // 280 x b128
      async_cp_b128(gx + i * 16, lx + i * 16);
    const char* gw = (const char*)wfrag;
    char* lw = (char*)wlds;
    for (int i = tid; i < 26 * 512 * 2 / 16; i += 256)     // 1664 x b128
      async_cp_b128(gw + i * 16, lw + i * 16);
  }
#else
  {
    const uint32_t* src = reinterpret_cast<const uint32_t*>(xpad + (size_t)n * XPAD_W);
    uint32_t* dst = reinterpret_cast<uint32_t*>(xrow);
    for (int i = tid; i < XPAD_W / 2; i += 256) dst[i] = src[i];
    const uint32_t* ws_ = reinterpret_cast<const uint32_t*>(wfrag);
    uint32_t* wd = reinterpret_cast<uint32_t*>(wlds);
    for (int i = tid; i < 26 * 512 / 2; i += 256) wd[i] = ws_[i];
  }
#endif
  for (int i = tid; i < 96; i += 256) { sg[i] = bn_g[i] * INV_BN; sb[i] = bn_b[i]; }
  if (tid < 32) s_lds[tid] = 0.0f;
#ifdef HAVE_ASYNC_LDS
  async_wait_all();
#endif
  __syncthreads();

  const int NCH[3]  = {2, 4, 7};      // padded K: 64, 128, 224
  const int OFFS[3] = {75, 50, 0};    // 100 - k/2
  const int FB[3]   = {0, 4, 12};
  const v16h* wf = reinterpret_cast<const v16h*>(wlds);

  float fsum[2][8];
  for (int m = 0; m < 2; ++m)
    for (int r = 0; r < 8; ++r) fsum[m][r] = 0.0f;

  for (int it = 0; it < 16; ++it) {
    int tile = wave + 8 * it;          // uniform per wave
    if (tile >= 126) break;
    int t0 = tile * 16;
    v8f zero = {};
    v8f acc[3][2];
    for (int s = 0; s < 3; ++s) { acc[s][0] = zero; acc[s][1] = zero; }

    for (int s = 0; s < 3; ++s) {
      int off = OFFS[s];
      for (int c = 0; c < NCH[s]; ++c) {
        v16h bf;
#pragma unroll
        for (int e = 0; e < 16; ++e) {
          int K = c * 32 + kmap(e, lane);
          bf[e] = xrow[off + t0 + tcol + K];
        }
#pragma unroll
        for (int m = 0; m < 2; ++m) {
          v16h af = wf[(size_t)(FB[s] + c * 2 + m) * 32 + lane];
          acc[s][m] = __builtin_amdgcn_wmma_f32_16x16x32_f16(
              false, af, false, bf, (short)0, acc[s][m], false, false);
        }
      }
    }
    int t = t0 + tcol;
    if (t < TPRIME) {
      for (int s = 0; s < 3; ++s)
        for (int m = 0; m < 2; ++m)
#pragma unroll
          for (int r = 0; r < 8; ++r) {
            int f = m * 16 + hi + r;
            float cv = acc[s][m][r] * sg[s * 32 + f] + sb[s * 32 + f];
            fsum[m][r] += geluf(cv);
          }
    }
  }
  for (int m = 0; m < 2; ++m)
    for (int r = 0; r < 8; ++r) {
      int f = m * 16 + hi + r;
      atomicAdd(&s_lds[f], fsum[m][r]);
    }
  __syncthreads();
  if (tid < 32) s_out[n * 32 + tid] = s_lds[tid] * (1.0f / (3.0f * (float)TPRIME));
}

// ---------------------------------------------------------------------------
// Spatial conv2d (kernel spans full C=64) + BN + GELU + time-sum (atomics).
// grid = B*16 blocks; block handles 8 consecutive 16-wide t tiles.
// xs tile staged async (strided rows, b32 granularity).
// ---------------------------------------------------------------------------
__global__ void __launch_bounds__(256) conv_sp_kernel(
    const _Float16* __restrict__ xpad, const _Float16* __restrict__ wfrag,
    const float* __restrict__ bns_g, const float* __restrict__ bns_b,
    float* __restrict__ mspat) {
  const int b  = blockIdx.x >> 4;
  const int tg = blockIdx.x & 15;
  const int tid = threadIdx.x, wave = tid >> 5, lane = tid & 31;
  const int tcol = lane & 15, hi = (lane & 16) ? 8 : 0;

  __shared__ __align__(16) _Float16 xs[64 * 174];   // row stride 174 halves (348 B, dword-aligned)
  __shared__ float acc_lds[32];

#ifdef HAVE_ASYNC_LDS
  {
    // per row: 174 halves = 87 dwords; global row base is 4-byte aligned
    for (int i = tid; i < 64 * 87; i += 256) {
      int c = i / 87, d = i % 87;
      const char* g = (const char*)(xpad + (size_t)(b * 64 + c) * XPAD_W + 90 + tg * 128) + d * 4;
      char* l = (char*)xs + ((size_t)c * 174) * 2 + d * 4;
      async_cp_b32(g, l);
    }
  }
#else
  for (int i = tid; i < 64 * 174; i += 256) {
    int c = i / 174, col = i % 174;
    xs[c * 174 + col] = xpad[(size_t)(b * 64 + c) * XPAD_W + 90 + tg * 128 + col];
  }
#endif
  if (tid < 32) acc_lds[tid] = 0.0f;
#ifdef HAVE_ASYNC_LDS
  async_wait_all();
#endif
  __syncthreads();

  const v16h* wf = reinterpret_cast<const v16h*>(wfrag);
  int t0 = tg * 128 + wave * 16;
  if (t0 < TPRIME) {                 // uniform per wave
    v8f zero = {};
    v8f acc[2] = {zero, zero};
    for (int c = 0; c < 64; ++c) {
      v16h bf;
#pragma unroll
      for (int e = 0; e < 16; ++e) bf[e] = xs[c * 174 + wave * 16 + tcol + kmap(e, lane)];
#pragma unroll
      for (int m = 0; m < 2; ++m) {
        v16h af = wf[(size_t)(c * 2 + m) * 32 + lane];
        acc[m] = __builtin_amdgcn_wmma_f32_16x16x32_f16(
            false, af, false, bf, (short)0, acc[m], false, false);
      }
    }
    int t = t0 + tcol;
    if (t < TPRIME) {
      for (int m = 0; m < 2; ++m)
#pragma unroll
        for (int r = 0; r < 8; ++r) {
          int o = m * 16 + hi + r;
          float v = acc[m][r] * INV_BN * bns_g[o] + bns_b[o];
          atomicAdd(&acc_lds[o], geluf(v));
        }
    }
  }
  __syncthreads();
  if (tid < 32) atomicAdd(&mspat[b * 32 + tid], acc_lds[tid]);
}

// ---------------------------------------------------------------------------
// SE1 gate: z = s * sigmoid(gelu(s@se1a^T)@se1b^T); also bp_feat.
// ---------------------------------------------------------------------------
__global__ void se1_kernel(const float* __restrict__ s, const float* __restrict__ se1a,
                           const float* __restrict__ se1b, const float* __restrict__ bp,
                           const float* __restrict__ bp_w, const float* __restrict__ bp_b,
                           float* __restrict__ z, float* __restrict__ bpf) {
  const int n = blockIdx.x, t = threadIdx.x;
  __shared__ float sr[32], hid[8];
  if (t < 32) sr[t] = s[n * 32 + t];
  __syncthreads();
  if (t < 8) {
    float a = 0.0f;
    for (int j = 0; j < 32; ++j) a += se1a[t * 32 + j] * sr[j];
    hid[t] = geluf(a);
  }
  __syncthreads();
  if (t < 32) {
    float a = 0.0f;
    for (int h = 0; h < 8; ++h) a += se1b[t * 8 + h] * hid[h];
    z[n * 32 + t] = sr[t] * sigmf(a);
  } else if (t < 36) {
    int d = t - 32;
    float a = bp_b[d];
    for (int j = 0; j < 4; ++j) a += bp[n * 4 + j] * bp_w[d * 4 + j];
    bpf[n * 4 + d] = a;
  }
}

// SE2 gate for spatial branch: v = mean * sigmoid(gelu(mean@se2a^T)@se2b^T)
__global__ void se2_kernel(const float* __restrict__ mspat, const float* __restrict__ se2a,
                           const float* __restrict__ se2b, float* __restrict__ vspat) {
  const int b = blockIdx.x, t = threadIdx.x;  // 32 threads
  __shared__ float m[32], hid[8];
  m[t] = mspat[b * 32 + t] * (1.0f / (float)TPRIME);
  __syncthreads();
  if (t < 8) {
    float a = 0.0f;
    for (int j = 0; j < 32; ++j) a += se2a[t * 32 + j] * m[j];
    hid[t] = geluf(a);
  }
  __syncthreads();
  float a = 0.0f;
  for (int h = 0; h < 8; ++h) a += se2b[t * 8 + h] * hid[h];
  vspat[b * 32 + t] = m[t] * sigmf(a);
}

// Concat [vspat(32) | z(32) | bpf(4)] then channel attention (ca1/ca2).
__global__ void feat_kernel(const float* __restrict__ vspat, const float* __restrict__ z,
                            const float* __restrict__ bpf, const float* __restrict__ ca1,
                            const float* __restrict__ ca2, float* __restrict__ feat) {
  const int n = blockIdx.x, b = n >> 6, t = threadIdx.x;  // 128 threads
  __shared__ float f0[68], hid[34];
  if (t < 32)       f0[t] = vspat[b * 32 + t];
  else if (t < 64)  f0[t] = z[n * 32 + (t - 32)];
  else if (t < 68)  f0[t] = bpf[n * 4 + (t - 64)];
  __syncthreads();
  if (t < 34) {
    float a = 0.0f;
    for (int j = 0; j < 68; ++j) a += ca1[t * 68 + j] * f0[j];
    hid[t] = geluf(a);
  }
  __syncthreads();
  if (t < 68) {
    float a = 0.0f;
    for (int h = 0; h < 34; ++h) a += ca2[t * 34 + h] * hid[h];
    feat[(size_t)n * 68 + t] = f0[t] * sigmf(a);
  }
}

// ---------------------------------------------------------------------------
// Generic WMMA f16 GEMM: C[M,N] = A[M,K] @ B. btrans=0: B is [K,N]; 1: B is [N,K].
// One wave per 16x16 C tile; fp32->f16 fragment gather; K zero-padded.
// ---------------------------------------------------------------------------
__global__ void gemm_wmma_kernel(const float* __restrict__ A, const float* __restrict__ B,
                                 float* __restrict__ C, int M, int N, int K, int btrans) {
  const int n0 = blockIdx.x * 16;
  const int m0 = blockIdx.y * 16;
  const int lane = threadIdx.x & 31;
  const int c16 = lane & 15;
  const int hi = (lane & 16) ? 8 : 0;
  v8f acc = {};
  const int KC = (K + 31) >> 5;
  for (int kc = 0; kc < KC; ++kc) {
    v16h af, bf;
    int arow = m0 + c16;
#pragma unroll
    for (int e = 0; e < 16; ++e) {
      int k = kc * 32 + kmap(e, lane);
      float av = (k < K) ? A[(size_t)arow * K + k] : 0.0f;
      float bv = 0.0f;
      if (k < K) bv = btrans ? B[(size_t)(n0 + c16) * K + k] : B[(size_t)k * N + (n0 + c16)];
      af[e] = (_Float16)av;
      bf[e] = (_Float16)bv;
    }
    acc = __builtin_amdgcn_wmma_f32_16x16x32_f16(false, af, false, bf, (short)0, acc, false, false);
  }
#pragma unroll
  for (int r = 0; r < 8; ++r)
    C[(size_t)(m0 + hi + r) * N + (n0 + c16)] = acc[r];
}

// Per-node attention logits: as[n,h] = sum_g H[n,h,g]*a_s[h,g] (and ad).
__global__ void attnprep_kernel(const float* __restrict__ H, const float* __restrict__ a_s,
                                const float* __restrict__ a_d, float* __restrict__ asb,
                                float* __restrict__ adb) {
  const int n = blockIdx.x, t = threadIdx.x;  // 128 threads
  __shared__ float red[128];
  for (int h = 0; h < 4; ++h) {
    float hv = H[(size_t)n * 512 + h * 128 + t];
    red[t] = hv * a_s[h * 128 + t];
    __syncthreads();
    for (int s = 64; s > 0; s >>= 1) { if (t < s) red[t] += red[t + s]; __syncthreads(); }
    if (t == 0) asb[n * 4 + h] = red[0];
    __syncthreads();
    red[t] = hv * a_d[h * 128 + t];
    __syncthreads();
    for (int s = 64; s > 0; s >>= 1) { if (t < s) red[t] += red[t + s]; __syncthreads(); }
    if (t == 0) adb[n * 4 + h] = red[0];
    __syncthreads();
  }
}

// GAT softmax+aggregate per (sample, head); fully-connected + self-loop dup.
// Adds 0.25*out (head mean) into outsum (pre-zeroed).
__global__ void __launch_bounds__(256) gat_agg_kernel(
    const float* __restrict__ H, const float* __restrict__ asb,
    const float* __restrict__ adb, float* __restrict__ outsum) {
  const int b = blockIdx.x >> 2, h = blockIdx.x & 3;
  const int t = threadIdx.x;  // 256 threads
  __shared__ float hsh[64][128];
  __shared__ float asv[64], adv[64], wj[64], scal[2];
  for (int i = t; i < 64 * 128; i += 256) {
    int j = i >> 7, g = i & 127;
    hsh[j][g] = H[(size_t)(b * 64 + j) * 512 + h * 128 + g];
  }
  if (t < 64) { asv[t] = asb[(b * 64 + t) * 4 + h]; adv[t] = adb[(b * 64 + t) * 4 + h]; }
  __syncthreads();
  for (int d = 0; d < 64; ++d) {
    if (t < 64) {
      float e = asv[t] + adv[d];
      wj[t] = (e > 0.0f) ? e : 0.2f * e;  // leaky_relu(0.2)
    }
    __syncthreads();
    if (t == 0) {
      float mx = wj[0];
      for (int j = 1; j < 64; ++j) mx = fmaxf(mx, wj[j]);
      scal[0] = mx;
    }
    __syncthreads();
    if (t < 64) wj[t] = expf(wj[t] - scal[0]);
    __syncthreads();
    if (t == 0) {
      float den = wj[d];  // self-loop duplicate
      for (int j = 0; j < 64; ++j) den += wj[j];
      scal[1] = den + 1e-16f;
    }
    __syncthreads();
    if (t < 128) {
      float num = wj[d] * hsh[d][t];
      for (int j = 0; j < 64; ++j) num += wj[j] * hsh[j][t];
      atomicAdd(&outsum[(size_t)(b * 64 + d) * 128 + t], 0.25f * num / scal[1]);
    }
    __syncthreads();
  }
}

// h = elu((h1+g1_b)*inv*bng_g + bng_b + skip + sk_b)
__global__ void combine1_kernel(const float* __restrict__ h1s, const float* __restrict__ g1_b,
                                const float* __restrict__ bng, const float* __restrict__ bnb,
                                const float* __restrict__ skv, const float* __restrict__ sk_b,
                                float* __restrict__ out) {
  int idx = blockIdx.x * blockDim.x + threadIdx.x;
  if (idx >= 1024 * 128) return;
  int g = idx & 127;
  float h1 = h1s[idx] + g1_b[g];
  out[idx] = eluf(h1 * INV_BN * bng[g] + bnb[g] + skv[idx] + sk_b[g]);
}

// h = elu((h2+g2_b)*inv*bng_g + bng_b + residual)
__global__ void combine2_kernel(const float* __restrict__ h2s, const float* __restrict__ g2_b,
                                const float* __restrict__ bng, const float* __restrict__ bnb,
                                const float* __restrict__ resid, float* __restrict__ out) {
  int idx = blockIdx.x * blockDim.x + threadIdx.x;
  if (idx >= 1024 * 128) return;
  int g = idx & 127;
  float h2 = h2s[idx] + g2_b[g];
  out[idx] = eluf(h2 * INV_BN * bng[g] + bnb[g] + resid[idx]);
}

// Mean-pool over nodes + 2-layer head.
__global__ void head_kernel(const float* __restrict__ hf, const float* __restrict__ c1w,
                            const float* __restrict__ c1b, const float* __restrict__ c2w,
                            const float* __restrict__ c2b, float* __restrict__ out) {
  const int b = blockIdx.x, t = threadIdx.x;  // 128 threads
  __shared__ float gm[128], hid[64];
  float a = 0.0f;
  for (int c = 0; c < 64; ++c) a += hf[(size_t)(b * 64 + c) * 128 + t];
  gm[t] = a * (1.0f / 64.0f);
  __syncthreads();
  if (t < 64) {
    float s = c1b[t];
    for (int g = 0; g < 128; ++g) s += c1w[t * 128 + g] * gm[g];
    hid[t] = geluf(s);
  }
  __syncthreads();
  if (t < 2) {
    float s = c2b[t];
    for (int hh = 0; hh < 64; ++hh) s += c2w[t * 64 + hh] * hid[hh];
    out[b * 2 + t] = s;
  }
}

// ---------------------------------------------------------------------------
extern "C" void kernel_launch(void* const* d_in, const int* in_sizes, int n_in,
                              void* d_out, int out_size, void* d_ws, size_t ws_size,
                              hipStream_t stream) {
  (void)in_sizes; (void)n_in; (void)out_size; (void)ws_size;
  // setup_inputs() dict order: x, bp, then params leaves in insertion order,
  // then edge_index, batch_idx (graph structure is known -> unused).
  const float* x      = (const float*)d_in[0];
  const float* bp     = (const float*)d_in[1];
  const float* w50    = (const float*)d_in[2];
  const float* w100   = (const float*)d_in[3];
  const float* w200   = (const float*)d_in[4];
  const float* bn_g   = (const float*)d_in[5];
  const float* bn_b   = (const float*)d_in[6];
  const float* se1a   = (const float*)d_in[7];
  const float* se1b   = (const float*)d_in[8];
  const float* w2d    = (const float*)d_in[9];
  const float* bns_g  = (const float*)d_in[10];
  const float* bns_b  = (const float*)d_in[11];
  const float* se2a   = (const float*)d_in[12];
  const float* se2b   = (const float*)d_in[13];
  const float* bp_w   = (const float*)d_in[14];
  const float* bp_b   = (const float*)d_in[15];
  const float* ca1    = (const float*)d_in[16];
  const float* ca2    = (const float*)d_in[17];
  const float* g1_w   = (const float*)d_in[18];
  const float* g1_as  = (const float*)d_in[19];
  const float* g1_ad  = (const float*)d_in[20];
  const float* g1_b   = (const float*)d_in[21];
  const float* g2_w   = (const float*)d_in[22];
  const float* g2_as  = (const float*)d_in[23];
  const float* g2_ad  = (const float*)d_in[24];
  const float* g2_b   = (const float*)d_in[25];
  const float* sk_w   = (const float*)d_in[26];
  const float* sk_b   = (const float*)d_in[27];
  const float* bng1_g = (const float*)d_in[28];
  const float* bng1_b = (const float*)d_in[29];
  const float* bng2_g = (const float*)d_in[30];
  const float* bng2_b = (const float*)d_in[31];
  const float* c1_w   = (const float*)d_in[32];
  const float* c1_b   = (const float*)d_in[33];
  const float* c2_w   = (const float*)d_in[34];
  const float* c2_b   = (const float*)d_in[35];
  float* out = (float*)d_out;

  char* ws = (char*)d_ws;
  auto alloc = [&](size_t bytes) -> char* {
    char* p = ws;
    ws += (bytes + 255) & ~(size_t)255;
    return p;
  };
  _Float16* xpad  = (_Float16*)alloc((size_t)1024 * XPAD_W * 2);
  _Float16* wms   = (_Float16*)alloc((size_t)26 * 512 * 2);
  _Float16* wsp   = (_Float16*)alloc((size_t)128 * 512 * 2);
  float* sbuf  = (float*)alloc((size_t)1024 * 32 * 4);
  float* zbuf  = (float*)alloc((size_t)1024 * 32 * 4);
  float* bpf   = (float*)alloc((size_t)1024 * 4 * 4);
  float* mspat = (float*)alloc((size_t)16 * 32 * 4);
  float* vspat = (float*)alloc((size_t)16 * 32 * 4);
  float* featb = (float*)alloc((size_t)1024 * 68 * 4);
  float* Hbuf  = (float*)alloc((size_t)1024 * 512 * 4);
  float* SKbuf = (float*)alloc((size_t)1024 * 128 * 4);
  float* h1sum = (float*)alloc((size_t)1024 * 128 * 4);
  float* hcur  = (float*)alloc((size_t)1024 * 128 * 4);
  float* h2sum = (float*)alloc((size_t)1024 * 128 * 4);
  float* hfin  = (float*)alloc((size_t)1024 * 128 * 4);
  float* asb   = (float*)alloc((size_t)1024 * 4 * 4);
  float* adb   = (float*)alloc((size_t)1024 * 4 * 4);

  (void)hipMemsetAsync(mspat, 0, (size_t)16 * 32 * 4, stream);
  (void)hipMemsetAsync(h1sum, 0, (size_t)1024 * 128 * 4, stream);
  (void)hipMemsetAsync(h2sum, 0, (size_t)1024 * 128 * 4, stream);

  pack_x_kernel<<<1024, 256, 0, stream>>>(x, xpad);
  pack_w_kernel<<<1, 256, 0, stream>>>(w50, w100, w200, w2d, wms, wsp);
  conv_ms_kernel<<<1024, 256, 0, stream>>>(xpad, wms, bn_g, bn_b, sbuf);
  conv_sp_kernel<<<256, 256, 0, stream>>>(xpad, wsp, bns_g, bns_b, mspat);
  se1_kernel<<<1024, 64, 0, stream>>>(sbuf, se1a, se1b, bp, bp_w, bp_b, zbuf, bpf);
  se2_kernel<<<16, 32, 0, stream>>>(mspat, se2a, se2b, vspat);
  feat_kernel<<<1024, 128, 0, stream>>>(vspat, zbuf, bpf, ca1, ca2, featb);

  // GAT layer 1
  gemm_wmma_kernel<<<dim3(32, 64), 32, 0, stream>>>(featb, g1_w, Hbuf, 1024, 512, 68, 0);
  attnprep_kernel<<<1024, 128, 0, stream>>>(Hbuf, g1_as, g1_ad, asb, adb);
  gemm_wmma_kernel<<<dim3(8, 64), 32, 0, stream>>>(featb, sk_w, SKbuf, 1024, 128, 68, 1);
  gat_agg_kernel<<<64, 256, 0, stream>>>(Hbuf, asb, adb, h1sum);
  combine1_kernel<<<512, 256, 0, stream>>>(h1sum, g1_b, bng1_g, bng1_b, SKbuf, sk_b, hcur);

  // GAT layer 2
  gemm_wmma_kernel<<<dim3(32, 64), 32, 0, stream>>>(hcur, g2_w, Hbuf, 1024, 512, 128, 0);
  attnprep_kernel<<<1024, 128, 0, stream>>>(Hbuf, g2_as, g2_ad, asb, adb);
  gat_agg_kernel<<<64, 256, 0, stream>>>(Hbuf, asb, adb, h2sum);
  combine2_kernel<<<512, 256, 0, stream>>>(h2sum, g2_b, bng2_g, bng2_b, hcur, hfin);

  head_kernel<<<16, 128, 0, stream>>>(hfin, c1_w, c1_b, c2_w, c2_b, out);
}
